// BasicBlock_2508260901478
// MI455X (gfx1250) — matrix operand order repaired and checked by reading, not compiled, not run on hardware
//
#include <hip/hip_runtime.h>

// ---------------------------------------------------------------------------
// MI455X (gfx1250) point-transformer BasicBlock.
// All GEMMs (10 dense + concat-fuse + 54 gathered conv taps) run through
// v_wmma_f32_16x16x32_f16 with f32 accumulate.
//  - LDS tiles XOR-swizzled: every WMMA fragment = two conflict-free
//    ds_load_b128 (all 64 banks covered).
//  - Conv gather is branch-free (pointer-select against a zeroed row) and
//    software-pipelined: tap k+1's gather/weights load into registers while
//    tap k's 32 WMMAs execute from LDS.
// ---------------------------------------------------------------------------

#define NPTS   160000
#define CCH    128
#define KNB    27
#define NSEG   2048
#define BN_EPS 1e-5f

typedef __attribute__((ext_vector_type(16))) _Float16 v16h;
typedef __attribute__((ext_vector_type(8)))  _Float16 v8h;
typedef __attribute__((ext_vector_type(8)))  float    v8f;

// LDS tile addressing: tile[row][k], 128x128 f16, 8-half (16B) chunks XOR-
// swizzled by row: chunk c of row r stored at (c ^ (r & 15)).
__device__ inline int sw_off(int row, int chunk) {
    return row * CCH + ((chunk ^ (row & 15)) << 3);
}

// A fragment (16x32 f16, ISA 7.12.2): lane half kb = (lane/16)*8 ;
// halfs 0..7 = K kk+kb.., halfs 8..15 = K kk+kb+16..  -> two 16B chunks.
__device__ inline v16h load_a_frag(const _Float16* As, int kk, int m0, int lane) {
    const int row = m0 + (lane & 15);
    const int c1  = (kk + ((lane >> 4) << 3)) >> 3;
    v8h lo = *(const v8h*)(As + sw_off(row, c1));
    v8h hi = *(const v8h*)(As + sw_off(row, c1 + 2));
    return __builtin_shufflevector(lo, hi, 0, 1, 2, 3, 4, 5, 6, 7,
                                   8, 9, 10, 11, 12, 13, 14, 15);
}

// B fragment (32x16 f16): same K order as A: halfs 0..7 = K kk+kb.., halfs 8..15 =
// K kk+kb+16.. (kb = (lane/16)*8) at col = n0 + lane%16 -> two 16B chunks of Bs[n][k].
__device__ inline v16h load_b_frag(const _Float16* Bs, int kk, int n0, int lane) {
    const int col = n0 + (lane & 15);
    const int c1  = (kk + ((lane >> 4) << 3)) >> 3;
    v8h lo = *(const v8h*)(Bs + sw_off(col, c1));
    v8h hi = *(const v8h*)(Bs + sw_off(col, c1 + 2));
    return __builtin_shufflevector(lo, hi, 0, 1, 2, 3, 4, 5, 6, 7,
                                   8, 9, 10, 11, 12, 13, 14, 15);
}

// stage one 128x128 f16 weight tile (pre-transposed to [n][k]) into LDS
__device__ inline void stage_b(_Float16* Bs, const _Float16* WhT, int tid) {
    const int n  = tid >> 1;
    const int k0 = (tid & 1) * 64;
    const _Float16* src = WhT + n * CCH + k0;
#pragma unroll
    for (int cj = 0; cj < 8; ++cj)
        *(v8h*)(Bs + sw_off(n, (k0 >> 3) + cj)) = *(const v8h*)(src + cj * 8);
}

// ---------------- Dense GEMM: Y[N,128] = [Xa | Xb][N,128*KCH] @ W ----------
// WhT holds KCH stacked 128x128 f16 weight tiles, each transposed to [n][k].
template <int KCH>
__global__ __launch_bounds__(256)
void gemm_wmma(const float* __restrict__ Xa, const float* __restrict__ Xb,
               const _Float16* __restrict__ WhT, float* __restrict__ Y) {
    __shared__ _Float16 As[CCH * CCH];
    __shared__ _Float16 Bs[CCH * CCH];
    const int tid  = threadIdx.x;
    const int lane = tid & 31;
    const int m0   = (tid >> 5) * 16;
    const long r0  = (long)blockIdx.x * 128;

    v8f acc[8] = {};

    for (int kc = 0; kc < KCH; ++kc) {
        if (kc) __syncthreads();
        // stage A: f32 rows -> f16, swizzled
        const float* X = (kc == 0) ? Xa : Xb;
        const int r  = tid >> 1;
        const int c0 = (tid & 1) * 64;
        const float* src = X + (r0 + r) * CCH + c0;
#pragma unroll
        for (int cj = 0; cj < 8; ++cj) {
            v8f xv = *(const v8f*)(src + cj * 8);
            *(v8h*)(As + sw_off(r, (c0 >> 3) + cj)) = __builtin_convertvector(xv, v8h);
        }
        stage_b(Bs, WhT + (long)kc * CCH * CCH, tid);
        __syncthreads();

#pragma unroll
        for (int kk2 = 0; kk2 < 4; ++kk2) {
            const int kk = kk2 * 32;
            v16h a = load_a_frag(As, kk, m0, lane);
#pragma unroll
            for (int t = 0; t < 8; ++t) {
                v16h b = load_b_frag(Bs, kk, t * 16, lane);
                acc[t] = __builtin_amdgcn_wmma_f32_16x16x32_f16(
                    false, a, false, b, (short)0, acc[t], false, false);
            }
        }
    }

    // C/D layout: n = lane%16 ; vgpr rr holds m = 8*(lane/16) + rr
    const int n  = lane & 15;
    const int mb = m0 + 8 * (lane >> 4);
#pragma unroll
    for (int t = 0; t < 8; ++t) {
        const float* ap = (const float*)&acc[t];
#pragma unroll
        for (int rr = 0; rr < 8; ++rr)
            Y[(r0 + mb + rr) * CCH + t * 16 + n] = ap[rr];
    }
}

// ---------------- Submanifold conv: Y = sum_k mask * gather(Xh) @ Wk --------
// Branch-free masked gather + register-level software pipeline over the taps.
__device__ inline void gather_regs(v8h* abuf, const _Float16* Xh,
                                   const _Float16* zrow, const int* nbr,
                                   const unsigned char* msk, long r0, int k,
                                   int tid) {
    const int r  = tid >> 1;
    const int c0 = (tid & 1) * 64;
    const long pr = (r0 + r) * (long)KNB + k;
    const int  gi = nbr[pr];
    const bool mv = msk[pr] != 0;
    const _Float16* src = mv ? (Xh + (long)gi * CCH + c0) : (zrow + c0);
#pragma unroll
    for (int cj = 0; cj < 8; ++cj)
        abuf[cj] = *(const v8h*)(src + cj * 8);
}

__device__ inline void wrow_regs(v8h* bbuf, const _Float16* WhT, int k, int tid) {
    const int n  = tid >> 1;
    const int k0 = (tid & 1) * 64;
    const _Float16* src = WhT + (long)k * CCH * CCH + n * CCH + k0;
#pragma unroll
    for (int cj = 0; cj < 8; ++cj)
        bbuf[cj] = *(const v8h*)(src + cj * 8);
}

__global__ __launch_bounds__(256)
void conv_wmma(const _Float16* __restrict__ Xh, const _Float16* __restrict__ WhT,
               const int* __restrict__ nbr, const unsigned char* __restrict__ msk,
               const _Float16* __restrict__ zrow, float* __restrict__ Y) {
    __shared__ _Float16 As[CCH * CCH];
    __shared__ _Float16 Bs[CCH * CCH];
    const int tid  = threadIdx.x;
    const int lane = tid & 31;
    const int m0   = (tid >> 5) * 16;
    const long r0  = (long)blockIdx.x * 128;

    v8f acc[8] = {};
    v8h abuf[8], bbuf[8];

    gather_regs(abuf, Xh, zrow, nbr, msk, r0, 0, tid);
    wrow_regs(bbuf, WhT, 0, tid);

    for (int k = 0; k < KNB; ++k) {
        // drain staging registers into swizzled LDS tiles
        const int r  = tid >> 1;
        const int c0 = (tid & 1) * 64;
#pragma unroll
        for (int cj = 0; cj < 8; ++cj) {
            *(v8h*)(As + sw_off(r, (c0 >> 3) + cj)) = abuf[cj];
            *(v8h*)(Bs + sw_off(r, (c0 >> 3) + cj)) = bbuf[cj];
        }
        __syncthreads();

        // issue next tap's global loads; latency hidden under 32 WMMAs
        if (k + 1 < KNB) {
            gather_regs(abuf, Xh, zrow, nbr, msk, r0, k + 1, tid);
            wrow_regs(bbuf, WhT, k + 1, tid);
        }

#pragma unroll
        for (int kk2 = 0; kk2 < 4; ++kk2) {
            const int kk = kk2 * 32;
            v16h a = load_a_frag(As, kk, m0, lane);
#pragma unroll
            for (int t = 0; t < 8; ++t) {
                v16h b = load_b_frag(Bs, kk, t * 16, lane);
                acc[t] = __builtin_amdgcn_wmma_f32_16x16x32_f16(
                    false, a, false, b, (short)0, acc[t], false, false);
            }
        }
        __syncthreads();
    }

    const int n  = lane & 15;
    const int mb = m0 + 8 * (lane >> 4);
#pragma unroll
    for (int t = 0; t < 8; ++t) {
        const float* ap = (const float*)&acc[t];
#pragma unroll
        for (int rr = 0; rr < 8; ++rr)
            Y[(r0 + mb + rr) * CCH + t * 16 + n] = ap[rr];
    }
}

// ---------------- Support kernels ------------------------------------------

__global__ void zero_f32(float* p, long n) {
    for (long i = (long)blockIdx.x * blockDim.x + threadIdx.x; i < n;
         i += (long)gridDim.x * blockDim.x)
        p[i] = 0.f;
}

__global__ void cvt_f16(const float* __restrict__ x, _Float16* __restrict__ y, long n) {
    for (long i = (long)blockIdx.x * blockDim.x + threadIdx.x; i < n;
         i += (long)gridDim.x * blockDim.x)
        y[i] = (_Float16)x[i];
}

// convert [nmat][128][128] f32 row-major ([k][n]) -> f16 transposed [n][k]
__global__ void cvt_f16_tr(const float* __restrict__ x, _Float16* __restrict__ y, long n) {
    for (long i = (long)blockIdx.x * blockDim.x + threadIdx.x; i < n;
         i += (long)gridDim.x * blockDim.x) {
        const long m  = i >> 14;
        const int  kk = (int)((i >> 7) & 127);
        const int  nn = (int)(i & 127);
        y[(m << 14) + (long)nn * CCH + kk] = (_Float16)x[i];
    }
}

// per-channel sum + sumsq (training-mode BN), coalesced rows + atomics
__global__ __launch_bounds__(128)
void colstats(const float* __restrict__ x, float* __restrict__ stats) {
    const int c   = threadIdx.x;
    const long r0 = (long)blockIdx.x * 128;
    float s = 0.f, s2 = 0.f;
    for (int r = 0; r < 128; ++r) {
        float v = x[(r0 + r) * CCH + c];
        s += v;
        s2 += v * v;
    }
    atomicAdd(&stats[c], s);
    atomicAdd(&stats[CCH + c], s2);
}

// mode 0: relu(bn) ; 1: relu(bn)*aux ; 2: relu(bn)+aux ; 3: relu(bn+aux)
__global__ void bn_apply(const float* __restrict__ x, float* __restrict__ out,
                         const float* __restrict__ stats,
                         const float* __restrict__ g, const float* __restrict__ b,
                         const float* __restrict__ aux, int mode, float invN, long n) {
    const long i = (long)blockIdx.x * blockDim.x + threadIdx.x;
    if (i >= n) return;
    const int c = (int)(i & (CCH - 1));
    const float m = stats[c] * invN;
    const float v = stats[CCH + c] * invN - m * m;
    const float z = (x[i] - m) * rsqrtf(v + BN_EPS) * g[c] + b[c];
    float o;
    if (mode == 0)      o = fmaxf(z, 0.f);
    else if (mode == 1) o = fmaxf(z, 0.f) * aux[i];
    else if (mode == 2) o = fmaxf(z, 0.f) + aux[i];
    else                o = fmaxf(z + aux[i], 0.f);
    out[i] = o;
}

__global__ void seg_sum_k(const float* __restrict__ x, const int* __restrict__ cl,
                          float* __restrict__ seg, float* __restrict__ cnt,
                          int withCnt, long n) {
    const long i = (long)blockIdx.x * blockDim.x + threadIdx.x;
    if (i >= n) return;
    const int p = (int)(i >> 7);
    const int c = (int)(i & (CCH - 1));
    const int s = cl[p];
    atomicAdd(&seg[(long)s * CCH + c], x[i]);
    if (withCnt && c == 0) atomicAdd(&cnt[s], 1.f);
}

__global__ void sub_seg_mean(float* __restrict__ x, const int* __restrict__ cl,
                             const float* __restrict__ seg, const float* __restrict__ cnt,
                             long n) {
    const long i = (long)blockIdx.x * blockDim.x + threadIdx.x;
    if (i >= n) return;
    const int p = (int)(i >> 7), c = (int)(i & (CCH - 1));
    const int s = cl[p];
    x[i] -= seg[(long)s * CCH + c] / fmaxf(cnt[s], 1.f);
}

__global__ void div_seg(float* __restrict__ x, const int* __restrict__ cl,
                        const float* __restrict__ seg, long n) {
    const long i = (long)blockIdx.x * blockDim.x + threadIdx.x;
    if (i >= n) return;
    const int p = (int)(i >> 7), c = (int)(i & (CCH - 1));
    x[i] /= (seg[(long)cl[p] * CCH + c] + 1e-6f);
}

__global__ void accum_fused(float* __restrict__ fused, const float* __restrict__ seg,
                            const int* __restrict__ cl, const float* __restrict__ adp,
                            int br, long n) {
    const long i = (long)blockIdx.x * blockDim.x + threadIdx.x;
    if (i >= n) return;
    const int p = (int)(i >> 7), c = (int)(i & (CCH - 1));
    fused[i] += adp[(long)p * 3 + br] * seg[(long)cl[p] * CCH + c];
}

__device__ inline unsigned fenc(float f) {
    unsigned u = __float_as_uint(f);
    return (u >> 31) ? ~u : (u | 0x80000000u);
}
__device__ inline float fdec(unsigned e) {
    unsigned u = (e >> 31) ? (e & 0x7FFFFFFFu) : ~e;
    return __uint_as_float(u);
}

__global__ void reduce_max_k(const float* __restrict__ x, long n, unsigned* gmax) {
    __shared__ float sd[256];
    float m = -3.4e38f;
    for (long i = (long)blockIdx.x * blockDim.x + threadIdx.x; i < n;
         i += (long)gridDim.x * blockDim.x)
        m = fmaxf(m, x[i]);
    sd[threadIdx.x] = m;
    __syncthreads();
    for (int s = 128; s > 0; s >>= 1) {
        if (threadIdx.x < s) sd[threadIdx.x] = fmaxf(sd[threadIdx.x], sd[threadIdx.x + s]);
        __syncthreads();
    }
    if (threadIdx.x == 0) atomicMax(gmax, fenc(sd[0]));
}

__global__ void exp_sub(float* __restrict__ x, const unsigned* gmax, long n) {
    const long i = (long)blockIdx.x * blockDim.x + threadIdx.x;
    if (i >= n) return;
    const float gm = fdec(*gmax);
    x[i] = expf(x[i] - gm);
}

__global__ void adaptive_k(const float* __restrict__ feat, const float* __restrict__ aW,
                           float* __restrict__ adp, int n) {
    const int i = blockIdx.x * blockDim.x + threadIdx.x;
    if (i >= n) return;
    float l0 = 0.f, l1 = 0.f, l2 = 0.f;
    const float* f = feat + (long)i * CCH;
    for (int c = 0; c < CCH; ++c) {
        const float v = f[c];
        l0 += v * aW[c * 3 + 0];
        l1 += v * aW[c * 3 + 1];
        l2 += v * aW[c * 3 + 2];
    }
    const float mx = fmaxf(l0, fmaxf(l1, l2));
    const float e0 = expf(l0 - mx), e1 = expf(l1 - mx), e2 = expf(l2 - mx);
    const float s = e0 + e1 + e2;
    adp[(long)i * 3 + 0] = e0 / s;
    adp[(long)i * 3 + 1] = e1 / s;
    adp[(long)i * 3 + 2] = e2 / s;
}

// ---------------- Orchestration --------------------------------------------

extern "C" void kernel_launch(void* const* d_in, const int* in_sizes, int n_in,
                              void* d_out, int out_size, void* d_ws, size_t ws_size,
                              hipStream_t stream) {
    (void)in_sizes; (void)n_in; (void)out_size; (void)ws_size;

    const long NC = (long)NPTS * CCH;
    const float invN = 1.0f / (float)NPTS;
    const int NB = NPTS / 128;               // 1250 row blocks
    const int EW = (int)((NC + 255) / 256);  // elementwise grid

    const float* feat      = (const float*)d_in[0];
    const int*   cl[3]     = {(const int*)d_in[1], (const int*)d_in[2], (const int*)d_in[3]};
    const int*   nbr       = (const int*)d_in[4];
    const unsigned char* msk = (const unsigned char*)d_in[5];
    const float* lw_W      = (const float*)d_in[6];
    const float* lw_gamma  = (const float*)d_in[7];
    const float* lw_beta   = (const float*)d_in[8];
    const float* w_W       = (const float*)d_in[9];
    const float* proj_W    = (const float*)d_in[10];
    const float* proj_gamma= (const float*)d_in[11];
    const float* proj_beta = (const float*)d_in[12];
    const float* adaptive_W= (const float*)d_in[13];
    const float* fuse_W    = (const float*)d_in[14];
    const float* fuse_gamma= (const float*)d_in[15];
    const float* fuse_beta = (const float*)d_in[16];
    const float* conv_W    = (const float*)d_in[17];
    const float* conv_gamma= (const float*)d_in[18];
    const float* conv_beta = (const float*)d_in[19];

    float* t0 = (float*)d_out;               // reuse output as a [N,C] temp

    // workspace layout
    float*     t1    = (float*)d_ws;                   // [N,C] f32
    float*     fbuf  = t1 + NC;                        // [N,C] f32
    _Float16*  xh    = (_Float16*)(fbuf + NC);         // [N,C] f16 (16B aligned)
    float*     adp   = (float*)(xh + NC);              // [N,3]
    float*     seg   = adp + (long)NPTS * 3;           // [2048,128]
    float*     cnt   = seg + (long)NSEG * CCH;         // [2048]
    float*     stats = cnt + NSEG;                     // [256]
    unsigned*  gmax  = (unsigned*)(stats + 256);       // [1] (+pad to 16 floats)
    float*     zrowf = stats + 256 + 16;               // 64 f32 = 128 f16 zeros
    _Float16*  zrow  = (_Float16*)zrowf;
    _Float16*  wh    = (_Float16*)(zrowf + 64);        // f16 weights, transposed

    const long WM = (long)CCH * CCH;                   // 16384
    _Float16* whLw   = wh;                             // 3 * WM
    _Float16* whW    = whLw + 3 * WM;                  // 3 * WM
    _Float16* whProj = whW + 3 * WM;                   // 4 * WM
    _Float16* whFuse = whProj + 4 * WM;                // 2 * WM (256x128)
    _Float16* whConv = whFuse + 2 * WM;                // 54 * WM

    // --- convert + transpose weights to f16 [n][k] once per call ---
    cvt_f16_tr<<<256, 256, 0, stream>>>(lw_W,   whLw,   3 * WM);
    cvt_f16_tr<<<256, 256, 0, stream>>>(w_W,    whW,    3 * WM);
    cvt_f16_tr<<<256, 256, 0, stream>>>(proj_W, whProj, 4 * WM);
    cvt_f16_tr<<<256, 256, 0, stream>>>(fuse_W, whFuse, 2 * WM);
    cvt_f16_tr<<<1024, 256, 0, stream>>>(conv_W, whConv, 54 * WM);
    zero_f32<<<1, 64, 0, stream>>>(zrowf, 64);

    // --- adaptive gates ---
    adaptive_k<<<(NPTS + 255) / 256, 256, 0, stream>>>(feat, adaptive_W, adp, NPTS);

    // --- fused-branch accumulator ---
    zero_f32<<<4096, 256, 0, stream>>>(fbuf, NC);

    // --- three cluster branches ---
    for (int i = 0; i < 3; ++i) {
        // pw = relu(bn(feat @ lw_W[i]))
        gemm_wmma<1><<<NB, 256, 0, stream>>>(feat, nullptr, whLw + i * WM, t0);
        zero_f32<<<1, 256, 0, stream>>>(stats, 256);
        colstats<<<NB, 128, 0, stream>>>(t0, stats);
        bn_apply<<<EW, 256, 0, stream>>>(t0, t0, stats, lw_gamma + i * CCH,
                                         lw_beta + i * CCH, nullptr, 0, invN, NC);
        // pw -= seg_mean(pw)[cl]
        zero_f32<<<1024, 256, 0, stream>>>(seg, (long)NSEG * CCH + NSEG);
        seg_sum_k<<<EW, 256, 0, stream>>>(t0, cl[i], seg, cnt, 1, NC);
        sub_seg_mean<<<EW, 256, 0, stream>>>(t0, cl[i], seg, cnt, NC);
        // pw = pw @ w_W[i] ; exp-normalize over clusters
        gemm_wmma<1><<<NB, 256, 0, stream>>>(t0, nullptr, whW + i * WM, t1);
        zero_f32<<<1, 32, 0, stream>>>((float*)gmax, 1);
        reduce_max_k<<<2048, 256, 0, stream>>>(t1, NC, gmax);
        exp_sub<<<EW, 256, 0, stream>>>(t1, gmax, NC);
        zero_f32<<<1024, 256, 0, stream>>>(seg, (long)NSEG * CCH);
        seg_sum_k<<<EW, 256, 0, stream>>>(t1, cl[i], seg, cnt, 0, NC);
        div_seg<<<EW, 256, 0, stream>>>(t1, cl[i], seg, NC);
        // pfeat = relu(bn(feat @ proj_W[i])) * pw ; cluster-sum ; gate-weight
        gemm_wmma<1><<<NB, 256, 0, stream>>>(feat, nullptr, whProj + i * WM, t0);
        zero_f32<<<1, 256, 0, stream>>>(stats, 256);
        colstats<<<NB, 128, 0, stream>>>(t0, stats);
        bn_apply<<<EW, 256, 0, stream>>>(t0, t0, stats, proj_gamma + i * CCH,
                                         proj_beta + i * CCH, t1, 1, invN, NC);
        zero_f32<<<1024, 256, 0, stream>>>(seg, (long)NSEG * CCH);
        seg_sum_k<<<EW, 256, 0, stream>>>(t0, cl[i], seg, cnt, 0, NC);
        accum_fused<<<EW, 256, 0, stream>>>(fbuf, seg, cl[i], adp, i, NC);
    }

    // --- f = relu(bn(concat[relu(bn(feat@proj3)), fused] @ fuse_W)) + feat ---
    gemm_wmma<1><<<NB, 256, 0, stream>>>(feat, nullptr, whProj + 3 * WM, t0);
    zero_f32<<<1, 256, 0, stream>>>(stats, 256);
    colstats<<<NB, 128, 0, stream>>>(t0, stats);
    bn_apply<<<EW, 256, 0, stream>>>(t0, t0, stats, proj_gamma + 3 * CCH,
                                     proj_beta + 3 * CCH, nullptr, 0, invN, NC);
    gemm_wmma<2><<<NB, 256, 0, stream>>>(t0, fbuf, whFuse, t1);
    zero_f32<<<1, 256, 0, stream>>>(stats, 256);
    colstats<<<NB, 128, 0, stream>>>(t1, stats);
    bn_apply<<<EW, 256, 0, stream>>>(t1, t0, stats, fuse_gamma, fuse_beta,
                                     feat, 2, invN, NC);           // t0 = f = res

    // --- conv1 -> bn+relu -> conv2 -> bn -> +res -> relu -------------------
    cvt_f16<<<4096, 256, 0, stream>>>(t0, xh, NC);
    conv_wmma<<<NB, 256, 0, stream>>>(xh, whConv, nbr, msk, zrow, fbuf);
    zero_f32<<<1, 256, 0, stream>>>(stats, 256);
    colstats<<<NB, 128, 0, stream>>>(fbuf, stats);
    bn_apply<<<EW, 256, 0, stream>>>(fbuf, fbuf, stats, conv_gamma, conv_beta,
                                     nullptr, 0, invN, NC);
    cvt_f16<<<4096, 256, 0, stream>>>(fbuf, xh, NC);
    conv_wmma<<<NB, 256, 0, stream>>>(xh, whConv + 27 * WM, nbr, msk, zrow, t1);
    zero_f32<<<1, 256, 0, stream>>>(stats, 256);
    colstats<<<NB, 128, 0, stream>>>(t1, stats);
    bn_apply<<<EW, 256, 0, stream>>>(t1, t0, stats, conv_gamma + CCH,
                                     conv_beta + CCH, t0, 3, invN, NC);
}
